// StackAttCore_5669356834704
// MI455X (gfx1250) — matrix operand order repaired and checked by reading, not compiled, not run on hardware
//
#include <hip/hip_runtime.h>
#include <hip/hip_bf16.h>
#include <math.h>

// ---------------------------------------------------------------------------
// StackAttCore for MI455X (gfx1250).
// GEMMs: v_wmma_f32_16x16x32_bf16, fp32->bf16 via v_cvt_pk_bf16_f32 during
// LDS staging, f32 accumulation. Block tile 128x128, wave tile 32x64.
// Software-pipelined double-buffered LDS; pipeline registers are individually
// named f4 scalars (no arrays, no lambdas -> no address-taken -> no scratch).
// ---------------------------------------------------------------------------

typedef __attribute__((ext_vector_type(16))) __bf16        v16bf;
typedef __attribute__((ext_vector_type(4)))  __bf16        v4bf;
typedef __attribute__((ext_vector_type(8)))  float         v8f;
typedef __attribute__((ext_vector_type(8)))  unsigned int  v8u;
typedef __attribute__((ext_vector_type(2)))  unsigned int  u2;
typedef __attribute__((ext_vector_type(4)))  float         f4;

#define B_ 256
#define L_ 196
#define R_ 1024
#define A_ 512
#define E_ 1024

#define BM 128
#define BN 128
#define BK 32
#define PADK 36   // LDS row stride (bf16): 4-elem granules stay 8B aligned
#define SPS 32    // steps per segment = K/BK = 1024/32

struct Seg { const float* A; int lda; const float* B; int ldb; };

__device__ __forceinline__ u2 cvt4_bf16(f4 v) {
  v4bf c = __builtin_convertvector(v, v4bf);   // v_cvt_pk_bf16_f32 x2
  return __builtin_bit_cast(u2, c);
}

__device__ __forceinline__ v8f wmma_bf16(v16bf a, v16bf b, v8f c) {
  return __builtin_amdgcn_wmma_f32_16x16x32_bf16(false, a, false, b,
                                                 (short)0, c, false, false);
}

__global__ __launch_bounds__(256) void wmma_gemm_kernel(
    Seg s0, Seg s1, Seg s2, int nseg,
    const float* __restrict__ bias0, const float* __restrict__ bias1,
    const float* __restrict__ cadd,
    float* __restrict__ out, int N)
{
  __shared__ unsigned short As[2][BM][PADK];
  __shared__ unsigned short Bs[2][BN][PADK];

  const int tid  = threadIdx.x;
  const int lane = tid & 31;
  const int wave = tid >> 5;
  const int lmod = lane & 15;
  const int lhi  = lane >> 4;
  const int mw   = (wave & 3) * 32;    // wave tile 32 (M) x 64 (N)
  const int nw   = (wave >> 2) * 64;
  const int bm0  = blockIdx.y * BM;
  const int bn0  = blockIdx.x * BN;

  // staging map: each thread owns 4 float4 granules of A and of B per tile
  const int srow = tid >> 3;           // 0..31
  const int scol = (tid & 7) * 4;      // 0..28

  v8f c00 = {}, c01 = {}, c02 = {}, c03 = {};
  v8f c10 = {}, c11 = {}, c12 = {}, c13 = {};

  const int total = nseg * SPS;

  // ---- prologue: issue loads for step 0 (seg 0, k0 = 0) ----
  f4 ra0, ra1, ra2, ra3, rb0, rb1, rb2, rb3;
  {
    const float* Ag = s0.A; const float* Bg = s0.B;
    const int lda = s0.lda, ldb = s0.ldb;
    ra0 = *(const f4*)&Ag[(long)(bm0 +  0 + srow) * lda + scol];
    ra1 = *(const f4*)&Ag[(long)(bm0 + 32 + srow) * lda + scol];
    ra2 = *(const f4*)&Ag[(long)(bm0 + 64 + srow) * lda + scol];
    ra3 = *(const f4*)&Ag[(long)(bm0 + 96 + srow) * lda + scol];
    rb0 = *(const f4*)&Bg[(long)(bn0 +  0 + srow) * ldb + scol];
    rb1 = *(const f4*)&Bg[(long)(bn0 + 32 + srow) * ldb + scol];
    rb2 = *(const f4*)&Bg[(long)(bn0 + 64 + srow) * ldb + scol];
    rb3 = *(const f4*)&Bg[(long)(bn0 + 96 + srow) * ldb + scol];
  }

  for (int s = 0; s < total; ++s) {
    const int buf = s & 1;

    // ---- stage: convert in-flight registers into LDS buffer `buf` ----
    *(u2*)&As[buf][ 0 + srow][scol] = cvt4_bf16(ra0);
    *(u2*)&As[buf][32 + srow][scol] = cvt4_bf16(ra1);
    *(u2*)&As[buf][64 + srow][scol] = cvt4_bf16(ra2);
    *(u2*)&As[buf][96 + srow][scol] = cvt4_bf16(ra3);
    *(u2*)&Bs[buf][ 0 + srow][scol] = cvt4_bf16(rb0);
    *(u2*)&Bs[buf][32 + srow][scol] = cvt4_bf16(rb1);
    *(u2*)&Bs[buf][64 + srow][scol] = cvt4_bf16(rb2);
    *(u2*)&Bs[buf][96 + srow][scol] = cvt4_bf16(rb3);
    __syncthreads();

    // ---- issue global loads for step s+1 (overlap with WMMA phase) ----
    if (s + 1 < total) {
      const int sn = s + 1;
      const int sg = sn >> 5;            // / SPS
      const int k0 = (sn & (SPS - 1)) * BK;
      const float* Ag = (sg == 0) ? s0.A   : (sg == 1) ? s1.A   : s2.A;
      const float* Bg = (sg == 0) ? s0.B   : (sg == 1) ? s1.B   : s2.B;
      const int   lda = (sg == 0) ? s0.lda : (sg == 1) ? s1.lda : s2.lda;
      const int   ldb = (sg == 0) ? s0.ldb : (sg == 1) ? s1.ldb : s2.ldb;
      ra0 = *(const f4*)&Ag[(long)(bm0 +  0 + srow) * lda + k0 + scol];
      ra1 = *(const f4*)&Ag[(long)(bm0 + 32 + srow) * lda + k0 + scol];
      ra2 = *(const f4*)&Ag[(long)(bm0 + 64 + srow) * lda + k0 + scol];
      ra3 = *(const f4*)&Ag[(long)(bm0 + 96 + srow) * lda + k0 + scol];
      rb0 = *(const f4*)&Bg[(long)(bn0 +  0 + srow) * ldb + k0 + scol];
      rb1 = *(const f4*)&Bg[(long)(bn0 + 32 + srow) * ldb + k0 + scol];
      rb2 = *(const f4*)&Bg[(long)(bn0 + 64 + srow) * ldb + k0 + scol];
      rb3 = *(const f4*)&Bg[(long)(bn0 + 96 + srow) * ldb + k0 + scol];
    }

    // ---- compute: fragments per documented CDNA5 wave32 layouts ----
    v8u au0, au1, bu0, bu1, bu2, bu3;
    #pragma unroll
    for (int v = 0; v < 8; ++v) {
      int ka = (v < 4) ? (2 * v + 8 * lhi) : (16 + 2 * (v - 4) + 8 * lhi);
      int kb = (v < 4) ? (2 * v + 8 * lhi) : (16 + 2 * (v - 4) + 8 * lhi);
      au0[v] = *(const unsigned int*)&As[buf][mw      + lmod][ka];
      au1[v] = *(const unsigned int*)&As[buf][mw + 16 + lmod][ka];
      bu0[v] = *(const unsigned int*)&Bs[buf][nw      + lmod][kb];
      bu1[v] = *(const unsigned int*)&Bs[buf][nw + 16 + lmod][kb];
      bu2[v] = *(const unsigned int*)&Bs[buf][nw + 32 + lmod][kb];
      bu3[v] = *(const unsigned int*)&Bs[buf][nw + 48 + lmod][kb];
    }
    v16bf a0 = __builtin_bit_cast(v16bf, au0);
    v16bf a1 = __builtin_bit_cast(v16bf, au1);
    v16bf b0 = __builtin_bit_cast(v16bf, bu0);
    v16bf b1 = __builtin_bit_cast(v16bf, bu1);
    v16bf b2 = __builtin_bit_cast(v16bf, bu2);
    v16bf b3 = __builtin_bit_cast(v16bf, bu3);

    c00 = wmma_bf16(a0, b0, c00);
    c01 = wmma_bf16(a0, b1, c01);
    c02 = wmma_bf16(a0, b2, c02);
    c03 = wmma_bf16(a0, b3, c03);
    c10 = wmma_bf16(a1, b0, c10);
    c11 = wmma_bf16(a1, b1, c11);
    c12 = wmma_bf16(a1, b2, c12);
    c13 = wmma_bf16(a1, b3, c13);
  }

  // ---- epilogue: C/D layout m = v + 8*lhi, n = lmod ----
  v8f accs[8] = { c00, c01, c02, c03, c10, c11, c12, c13 };
  #pragma unroll
  for (int fm = 0; fm < 2; ++fm) {
    #pragma unroll
    for (int fn = 0; fn < 4; ++fn) {
      v8f acc = accs[fm * 4 + fn];
      #pragma unroll
      for (int v = 0; v < 8; ++v) {
        int m = bm0 + mw + fm * 16 + 8 * lhi + v;
        int n = bn0 + nw + fn * 16 + lmod;
        float r = acc[v];
        if (bias0) r += bias0[n];
        if (bias1) r += bias1[n];
        if (cadd)  r += cadd[(long)m * N + n];
        out[(long)m * N + n] = r;
      }
    }
  }
}

// LSTM gate nonlinearity: S is (B, 5R); maxout in_transform; write h,c.
__global__ __launch_bounds__(256) void lstm_point_kernel(
    const float* __restrict__ S, const float* __restrict__ c_prev,
    float* __restrict__ h_out, float* __restrict__ c_out,
    float* __restrict__ h_dup)
{
  int i = blockIdx.x * 256 + threadIdx.x;
  int b = i >> 10, r = i & (R_ - 1);
  const float* row = S + (long)b * (5 * R_);
  float si = row[r], sf = row[R_ + r], so = row[2 * R_ + r];
  float t  = fmaxf(row[3 * R_ + r], row[4 * R_ + r]);
  float ig = 1.f / (1.f + __expf(-si));
  float fg = 1.f / (1.f + __expf(-sf));
  float og = 1.f / (1.f + __expf(-so));
  float c  = fg * c_prev[i] + ig * t;
  float h  = og * tanhf(c);
  h_out[i] = h;
  c_out[i] = c;
  if (h_dup) h_dup[i] = h;
}

// scores[b,l] = ab + sum_a tanh(p_att[b,l,a] + att_h[b,a]) * aw[a]
__global__ __launch_bounds__(256) void att_scores_kernel(
    const float* __restrict__ att_h,   // (B, A)
    const float* __restrict__ p_att,   // (B, L, A)
    const float* __restrict__ aw,      // (A)
    const float* __restrict__ ab,      // (1)
    float* __restrict__ scores)        // (B, L)
{
  __shared__ float hsh[A_];
  int b = blockIdx.x;
  for (int i = threadIdx.x; i < A_; i += 256) hsh[i] = att_h[(long)b * A_ + i];
  __syncthreads();

  int wave = threadIdx.x >> 5, lane = threadIdx.x & 31;
  int l = blockIdx.y * 8 + wave;
  if (l < L_) {
    const float* p = p_att + ((long)b * L_ + l) * A_;
    float sum = 0.f;
    #pragma unroll 4
    for (int a = lane; a < A_; a += 32)
      sum += tanhf(p[a] + hsh[a]) * aw[a];
    #pragma unroll
    for (int off = 16; off; off >>= 1)
      sum += __shfl_xor(sum, off, 32);
    if (lane == 0) scores[(long)b * L_ + l] = sum + ab[0];
  }
}

// softmax over L then context[b,d] = sum_l w[b,l] * att_feats[b,l,d]
__global__ __launch_bounds__(256) void att_apply_kernel(
    const float* __restrict__ scores,     // (B, L)
    const float* __restrict__ att_feats,  // (B, L, R)
    float* __restrict__ ctx)              // (B, R)
{
  __shared__ float w[L_];
  __shared__ float red[256];
  int b = blockIdx.x, tid = threadIdx.x;

  float s = (tid < L_) ? scores[(long)b * L_ + tid] : -1e30f;
  red[tid] = s; __syncthreads();
  #pragma unroll
  for (int o = 128; o; o >>= 1) {
    if (tid < o) red[tid] = fmaxf(red[tid], red[tid + o]);
    __syncthreads();
  }
  float mx = red[0]; __syncthreads();

  float e = (tid < L_) ? __expf(s - mx) : 0.f;
  red[tid] = e; __syncthreads();
  #pragma unroll
  for (int o = 128; o; o >>= 1) {
    if (tid < o) red[tid] += red[tid + o];
    __syncthreads();
  }
  float inv = 1.f / red[0];
  if (tid < L_) w[tid] = e * inv;
  __syncthreads();

  const float* af = att_feats + (long)b * L_ * R_;
  for (int d = tid; d < R_; d += 256) {
    float a0 = 0.f, a1 = 0.f;
    #pragma unroll 2
    for (int l = 0; l < L_; l += 2) {
      a0 += w[l] * af[(long)l * R_ + d];
      if (l + 1 < L_) a1 += w[l + 1] * af[(long)(l + 1) * R_ + d];
    }
    ctx[(long)b * R_ + d] = a0 + a1;
  }
}

extern "C" void kernel_launch(void* const* d_in, const int* in_sizes, int n_in,
                              void* d_out, int out_size, void* d_ws, size_t ws_size,
                              hipStream_t stream) {
  const float* xt       = (const float*)d_in[0];
  const float* fc_feats = (const float*)d_in[1];
  const float* att_f    = (const float*)d_in[2];
  const float* p_att    = (const float*)d_in[3];
  const float* state_h  = (const float*)d_in[4];
  const float* state_c  = (const float*)d_in[5];
  const float* l0_wi = (const float*)d_in[6];  const float* l0_bi = (const float*)d_in[7];
  const float* l0_wh = (const float*)d_in[8];  const float* l0_bh = (const float*)d_in[9];
  const float* l1_wi = (const float*)d_in[10]; const float* l1_bi = (const float*)d_in[11];
  const float* l1_wh = (const float*)d_in[12]; const float* l1_bh = (const float*)d_in[13];
  const float* l2_wi = (const float*)d_in[14]; const float* l2_bi = (const float*)d_in[15];
  const float* l2_wh = (const float*)d_in[16]; const float* l2_bh = (const float*)d_in[17];
  const float* a1_hw = (const float*)d_in[18]; const float* a1_hb = (const float*)d_in[19];
  const float* a1_aw = (const float*)d_in[20]; const float* a1_ab = (const float*)d_in[21];
  const float* a2_hw = (const float*)d_in[22]; const float* a2_hb = (const float*)d_in[23];
  const float* a2_aw = (const float*)d_in[24]; const float* a2_ab = (const float*)d_in[25];
  const float* emb2_w = (const float*)d_in[26]; const float* emb2_b = (const float*)d_in[27];

  float* out = (float*)d_out;
  float* ws  = (float*)d_ws;

  const long BR = (long)B_ * R_;           // 262144
  float* S      = ws;                      // B x 5R
  float* attH   = S      + (long)B_ * 5 * R_;
  float* scores = attH   + (long)B_ * A_;
  float* att1   = scores + (long)B_ * L_;
  float* att2   = att1   + BR;
  float* hprime = att2   + BR;

  // d_out layout: [h2 | new_h(3,B,R) | new_c(3,B,R)]
  float* h0  = out + BR * 1;
  float* h1  = out + BR * 2;
  float* h2  = out + BR * 3;
  float* c0  = out + BR * 4;
  float* c1  = out + BR * 5;
  float* c2  = out + BR * 6;

  dim3 blk(256);
  Seg z{nullptr, 0, nullptr, 0};
  auto gemm = [&](Seg a, Seg b, Seg c, int nseg,
                  const float* bi0, const float* bi1, const float* cadd,
                  float* o, int M, int N) {
    dim3 grid(N / BN, M / BM);
    wmma_gemm_kernel<<<grid, blk, 0, stream>>>(a, b, c, nseg,
                                               bi0, bi1, cadd, o, N);
  };

  // ---- LSTM layer 0: S = [xt|fc|h_prev0] @ [Wi|Wh]^T + bi + bh ----
  gemm(Seg{xt, E_, l0_wi, 2048}, Seg{fc_feats, R_, l0_wi + 1024, 2048},
       Seg{state_h, R_, l0_wh, R_}, 3, l0_bi, l0_bh, nullptr, S, B_, 5 * R_);
  lstm_point_kernel<<<(B_ * R_) / 256, blk, 0, stream>>>(S, state_c, h0, c0, nullptr);

  // ---- attention 1 ----
  gemm(Seg{h0, R_, a1_hw, R_}, z, z, 1, a1_hb, nullptr, nullptr, attH, B_, A_);
  att_scores_kernel<<<dim3(B_, (L_ + 7) / 8), blk, 0, stream>>>(attH, p_att, a1_aw, a1_ab, scores);
  att_apply_kernel<<<B_, blk, 0, stream>>>(scores, att_f, att1);

  // ---- LSTM layer 1 ----
  gemm(Seg{h0, R_, l1_wi, 2048}, Seg{att1, R_, l1_wi + 1024, 2048},
       Seg{state_h + BR, R_, l1_wh, R_}, 3, l1_bi, l1_bh, nullptr, S, B_, 5 * R_);
  lstm_point_kernel<<<(B_ * R_) / 256, blk, 0, stream>>>(S, state_c + BR, h1, c1, nullptr);

  // ---- h' = h1 + att1 @ emb2_w^T + emb2_b ----
  gemm(Seg{att1, R_, emb2_w, R_}, z, z, 1, emb2_b, nullptr, h1, hprime, B_, R_);

  // ---- attention 2 ----
  gemm(Seg{hprime, R_, a2_hw, R_}, z, z, 1, a2_hb, nullptr, nullptr, attH, B_, A_);
  att_scores_kernel<<<dim3(B_, (L_ + 7) / 8), blk, 0, stream>>>(attH, p_att, a2_aw, a2_ab, scores);
  att_apply_kernel<<<B_, blk, 0, stream>>>(scores, att_f, att2);

  // ---- LSTM layer 2 (h2 duplicated into out[0:BR]) ----
  gemm(Seg{h1, R_, l2_wi, 2048}, Seg{att2, R_, l2_wi + 1024, 2048},
       Seg{state_h + 2 * BR, R_, l2_wh, R_}, 3, l2_bi, l2_bh, nullptr, S, B_, 5 * R_);
  lstm_point_kernel<<<(B_ * R_) / 256, blk, 0, stream>>>(S, state_c + 2 * BR, h2, c2, out);
}
